// Model_38689065402600
// MI455X (gfx1250) — matrix operand order repaired and checked by reading, not compiled, not run on hardware
//
#include <hip/hip_runtime.h>
#include <hip/hip_bf16.h>
#include <math.h>

// ---------------------------------------------------------------------------
// out[b] = silu( A · Wfold ) · fc_w + fc_b
//   A[b,(k,slot)] = G[b,slot] * f[b,i,k] * f[b,j,k]   (built on the fly, bf16)
//   Wfold[(k,slot),o] = sign/symmetry-folded W        (precomputed, bf16)
// GEMM: [4096 x 47104] · [47104 x 64], v_wmma_f32_16x16x32_bf16, f32 accum.
//
// Slot mapping per k (736 slots = 23 chunks of 32):
//   run i (i=0..35) occupies slots [R_i, R_i+len_i), len_i = 36-(i&~3),
//   slot offset o -> j = (i&~3)+o.  Valid pair iff j>=i; other slots are
//   dead (Wfold==0, A finite).  Slots 720..735 are tail padding (Wfold==0).
// ---------------------------------------------------------------------------

typedef __attribute__((ext_vector_type(16))) __bf16 v16bf;
typedef __attribute__((ext_vector_type(4)))  __bf16 v4bf;
typedef __attribute__((ext_vector_type(8)))  float  v8f;

#define NDIR    36
#define NSLOT   736              // 720 run slots + 16 pad (23*32)
#define NCH     23               // K-chunks of 32 per k
#define NK      64
#define NGRP    184              // 736/4 groups per batch row
#define FOLD_TOTAL (NK * NCH * 4 * 32 * 16)   // bf16 elems, fragment-linear

__device__ inline void dirDecode(int i, int& A, int& B) {
  int a = 0, r = i;
  while (r >= 8 - a) { r -= 8 - a; ++a; }
  A = a; B = a + 1 + r;
}

// slot t (0..735) -> packed (i | j<<8), or 0 for dead/pad slots
__device__ inline unsigned int slotIJ(int t) {
  if (t >= 720) return 0u;
  int R = 0;
  for (int i = 0; i < 36; ++i) {
    int len = 36 - (i & ~3);
    if (t < R + len) {
      int j = (i & ~3) + (t - R);
      return (j >= i) ? (unsigned int)(i | (j << 8)) : 0u;
    }
    R += len;
  }
  return 0u;
}

// ---------------------------------------------------------------------------
// Preprocess: fold W[9,9,9,9,64,64] -> Wfold bf16, WMMA-B-fragment-linear:
//   Wfold[(((k*23 + c)*4 + otile)*32 + lane)*16 + e]
// lane holds column N=lane%16; elements e cover K-halves 0..15 / 16..31.
// ---------------------------------------------------------------------------
__global__ __launch_bounds__(256) void fold_w_kernel(
    const float* __restrict__ W, __bf16* __restrict__ wf)
{
  int idx = blockIdx.x * 256 + threadIdx.x;
  if (idx >= FOLD_TOTAL) return;
  int e     =  idx        & 15;
  int lane  = (idx >> 4)  & 31;
  int otile = (idx >> 9)  & 3;
  int kc    =  idx >> 11;            // k*23 + c
  int k = kc / NCH;
  int c = kc % NCH;
  int tl = ((e < 8) ? e : e + 8) + ((lane < 16) ? 0 : 8);
  int t  = c * 32 + tl;
  float val = 0.f;
  unsigned int ij = slotIJ(t);
  if (ij != 0u || t == 0) {          // t==0 is the valid pair (0,0)
    int i = ij & 0xff, j = ij >> 8;
    if (j >= i) {                    // always true for encoded slots
      int A, B;  dirDecode(i, A, B);
      int Dd, E; dirDecode(j, Dd, E);
      int ko = k * 64 + otile * 16 + (lane & 15);
      #define W4(a,b,d,ee) W[(size_t)((((a)*9+(b))*9+(d))*9+(ee)) * 4096 + ko]
      val = W4(A,B,Dd,E) - W4(A,B,E,Dd) - W4(B,A,Dd,E) + W4(B,A,E,Dd);
      if (i != j)
        val += W4(Dd,E,A,B) - W4(Dd,E,B,A) - W4(E,Dd,A,B) + W4(E,Dd,B,A);
      #undef W4
    }
  }
  wf[idx] = (__bf16)val;
}

// ---------------------------------------------------------------------------
// Main kernel: 1 block = 16 batch rows, 4 wave32s = 4 output tiles of 16.
// ---------------------------------------------------------------------------
__global__ __launch_bounds__(128) void tensor_net_kernel(
    const float* __restrict__ x,     const float* __restrict__ means,
    const float* __restrict__ betas, const float* __restrict__ as_w,
    const float* __restrict__ as_b,  const float* __restrict__ fc_w,
    const float* __restrict__ fc_b,  const __bf16* __restrict__ wf,
    float* __restrict__ out)
{
  // aliased region: setup-only arrays overlap the main-loop A panel + C_red
  __shared__ __align__(16) unsigned char u_lds[27648];
  float* x_lds  = (float*)u_lds;                // 432 f32   (setup)
  float* dn_lds = (float*)(u_lds + 1728);       // 1728 f32  (setup)
  float* asw    = (float*)(u_lds + 8640);       // 3200 f32  (setup)
  float* mb     = (float*)(u_lds + 21440);      // 100 f32   (setup)
  __bf16* Apan  = (__bf16*)u_lds;               // 16x736 bf16 (main loop)
  float* C_red  = (float*)(u_lds + 23552);      // 16x64 f32   (epilogue)

  __shared__ __bf16 f_lds[16 * NK * NDIR];      // f[m][k][i]     73728 B
  __shared__ float  G_lds[16 * NSLOT];          // G[m][slot]     47104 B
  __shared__ unsigned short tIJ[NSLOT];         // slot -> (i,j)   1472 B
  __shared__ unsigned int   sched[NGRP];        // group descrip.   736 B
  __shared__ unsigned char  dA[NDIR], dB[NDIR];

  const int tid   = threadIdx.x;
  const int lane  = tid & 31;
  const int wave  = tid >> 5;                   // output tile
  const int bbase = blockIdx.x * 16;

  // ---- stage inputs & build tables ----
  for (int idx = tid; idx < 16 * 27; idx += 128)
    x_lds[idx] = x[(size_t)bbase * 27 + idx];
  for (int idx = tid; idx < NK * 50; idx += 128)
    asw[idx] = as_w[idx];
  if (tid < 50) { mb[tid] = means[tid]; mb[50 + tid] = betas[tid]; }
  for (int i = tid; i < NDIR; i += 128) {
    int A, B; dirDecode(i, A, B);
    dA[i] = (unsigned char)A; dB[i] = (unsigned char)B;
  }
  for (int t = tid; t < NSLOT; t += 128)
    tIJ[t] = (unsigned short)slotIJ(t);
  if (tid == 0) {                               // group schedule (once)
    int idx = 0, R = 0;
    for (int i = 0; i < 36; ++i) {
      int jb0 = i & ~3, len = 36 - jb0;
      for (int gi = 0; gi < len / 4; ++gi)
        sched[idx++] = (unsigned int)((R + 4 * gi) | ((jb0 + 4 * gi) << 10) | (i << 16));
      R += len;
    }
    for (int p = 0; p < 4; ++p)                 // tail-pad groups
      sched[idx++] = (unsigned int)(720 + 4 * p);
  }
  __syncthreads();

  // ---- per (batch-row, direction) geometry + RBF features ----
  for (int task = tid; task < 16 * NDIR; task += 128) {
    int m = task / NDIR, i = task % NDIR;
    int A = dA[i], B = dB[i];
    float dx = x_lds[(m * 9 + A) * 3 + 0] - x_lds[(m * 9 + B) * 3 + 0];
    float dy = x_lds[(m * 9 + A) * 3 + 1] - x_lds[(m * 9 + B) * 3 + 1];
    float dz = x_lds[(m * 9 + A) * 3 + 2] - x_lds[(m * 9 + B) * 3 + 2];
    float d2  = dx * dx + dy * dy + dz * dz;
    float inv = 1.f / (d2 + 1e-6f);
    dn_lds[(m * NDIR + i) * 3 + 0] = dx * inv;
    dn_lds[(m * NDIR + i) * 3 + 1] = dy * inv;
    dn_lds[(m * NDIR + i) * 3 + 2] = dz * inv;
    // ExpNormalSmearing of SQUARED distance (faithful to reference)
    float cut = (d2 < 5.f) ? 0.5f * (cosf(d2 * 0.6283185307179586f) + 1.f) : 0.f;
    float ein = expf(-d2);                       // ALPHA==1, CUTOFF_LOWER==0
    float rbf[50];
    #pragma unroll
    for (int r = 0; r < 50; ++r) {
      float dlt = ein - mb[r];
      rbf[r] = cut * expf(-mb[50 + r] * dlt * dlt);
    }
    for (int k = 0; k < NK; ++k) {
      float acc = as_b[k];
      #pragma unroll
      for (int r = 0; r < 50; ++r) acc += rbf[r] * asw[k * 50 + r];
      f_lds[(m * NK + k) * NDIR + i] = (__bf16)acc;
    }
  }
  __syncthreads();

  // ---- Gram matrix (dead/pad slots get finite dn0.dn0 -- killed by Wfold=0)
  for (int m = 0; m < 16; ++m)
    for (int t = tid; t < NSLOT; t += 128) {
      unsigned int ij = tIJ[t];
      int i = ij & 0xff, j = ij >> 8;
      const float* a = &dn_lds[(m * NDIR + i) * 3];
      const float* b = &dn_lds[(m * NDIR + j) * 3];
      G_lds[m * NSLOT + t] = a[0] * b[0] + a[1] * b[1] + a[2] * b[2];
    }
  __syncthreads();

  // ---- main GEMM: 64 rounds (one k each) of 23 x wmma_f32_16x16x32_bf16 ----
  unsigned int sc[NCH];                          // this thread's 23 groups
  #pragma unroll
  for (int g = 0; g < NCH; ++g) sc[g] = sched[(tid & 7) + 8 * g];
  const int m    = tid >> 3;                     // 8 builder threads per row
  const int arow = (lane & 15) * (NSLOT * 2);    // A-panel row stride bytes
  const int half = (lane < 16) ? 0 : 16;

  v8f acc = {};
  for (int k = 0; k < NK; ++k) {
    const __bf16* fb = f_lds + (m * NK + k) * NDIR;
    const float*  Gb = G_lds + m * NSLOT;
    __bf16*       Ab = Apan  + m * NSLOT;
    #pragma unroll
    for (int g = 0; g < NCH; ++g) {              // build 4 slots per group
      unsigned int s = sc[g];
      int tb = s & 1023, jb = (s >> 10) & 63, ii = s >> 16;
      float  fi = (float)fb[ii];                 // scalar, hoisted per run
      v4bf   fj = *(const v4bf*)(fb + jb);       // ds_load_b64 (aligned)
      float4 g4 = *(const float4*)(Gb + tb);     // ds_load_b128
      v4bf o;
      o[0] = (__bf16)(g4.x * fi * (float)fj[0]);
      o[1] = (__bf16)(g4.y * fi * (float)fj[1]);
      o[2] = (__bf16)(g4.z * fi * (float)fj[2]);
      o[3] = (__bf16)(g4.w * fi * (float)fj[3]);
      *(v4bf*)(Ab + tb) = o;                     // ds_store_b64
    }
    __syncthreads();

    const __bf16* wbase = wf + ((size_t)(k * NCH) * 4 + wave) * 512 + lane * 16;
    if (k + 1 < NK)                              // global_prefetch next k's B
      __builtin_prefetch(wf + ((size_t)((k + 1) * NCH) * 4 + wave) * 512 + lane * 16, 0, 0);

    #pragma unroll
    for (int c = 0; c < NCH; ++c) {
      // A fragment per CDNA5 16-bit A layout: M=lane%16,
      // lanes<16: K 0..7 & 16..23 ; lanes>=16: K 8..15 & 24..31
      union { uint4 q[2]; v16bf v; } ua;
      const char* ab = (const char*)Apan + arow + c * 64;
      ua.q[0] = *(const uint4*)(ab + half);
      ua.q[1] = *(const uint4*)(ab + 32 + half);
      v16bf bv = *(const v16bf*)(wbase + (size_t)c * 2048);
      acc = __builtin_amdgcn_wmma_f32_16x16x32_bf16(
          false, ua.v, false, bv, (short)0, acc, false, false);
    }
    __syncthreads();
  }

  // ---- epilogue: SiLU + fc reduction ----
  #pragma unroll
  for (int r = 0; r < 8; ++r) {
    int row = r + ((lane >= 16) ? 8 : 0);
    int col = wave * 16 + (lane & 15);
    C_red[row * 64 + col] = acc[r];
  }
  __syncthreads();
  if (tid < 16) {
    float s = fc_b[0];
    for (int o = 0; o < 64; ++o) {
      float v = C_red[tid * 64 + o];
      s += (v / (1.f + expf(-v))) * fc_w[o];
    }
    out[bbase + tid] = s;
  }
}

extern "C" void kernel_launch(void* const* d_in, const int* in_sizes, int n_in,
                              void* d_out, int out_size, void* d_ws, size_t ws_size,
                              hipStream_t stream) {
  const float* x     = (const float*)d_in[0];
  const float* W     = (const float*)d_in[1];
  const float* means = (const float*)d_in[2];
  const float* betas = (const float*)d_in[3];
  const float* as_w  = (const float*)d_in[4];
  const float* as_b  = (const float*)d_in[5];
  const float* fc_w  = (const float*)d_in[6];
  const float* fc_b  = (const float*)d_in[7];
  float* out = (float*)d_out;
  __bf16* wf = (__bf16*)d_ws;   // needs FOLD_TOTAL*2 = 6,029,312 bytes

  // 1) fold W (reads the 107 MB tensor once -- HBM roofline floor ~4.6 us)
  fold_w_kernel<<<(FOLD_TOTAL + 255) / 256, 256, 0, stream>>>(W, wf);
  // 2) fused feature + WMMA GEMM + SiLU/fc head
  tensor_net_kernel<<<4096 / 16, 128, 0, stream>>>(
      x, means, betas, as_w, as_b, fc_w, fc_b, wf, out);
}
